// FeatExtractionEarlyFusion_65953517797620
// MI455X (gfx1250) — matrix operand order repaired and checked by reading, not compiled, not run on hardware
//
#include <hip/hip_runtime.h>
#include <hip/hip_bf16.h>
#include <math.h>

typedef __attribute__((ext_vector_type(16))) _Float16 v16h;
typedef __attribute__((ext_vector_type(8)))  _Float16 v8h;
typedef __attribute__((ext_vector_type(4)))  _Float16 v4h;
typedef __attribute__((ext_vector_type(8)))  float    v8f;

#define NPTS   2048
#define BATCH  2
#define NSAMP  64
#define RAD2   0.09f

static constexpr size_t PP      = (size_t)BATCH * NPTS * NSAMP;  // 262144 positions (pre layers)
static constexpr size_t PPB_PRE = (size_t)NPTS * NSAMP;          // 131072 per batch
static constexpr size_t PP2     = (size_t)BATCH * NPTS;          // 4096 positions (post layers)

// ---------------------------------------------------------------------------
// f32 -> f16 weight conversion (run once per launch; tiny)
// ---------------------------------------------------------------------------
__global__ void k_cvt(const float* __restrict__ s, _Float16* __restrict__ d, int n) {
  int i = blockIdx.x * blockDim.x + threadIdx.x;
  if (i < n) d[i] = (_Float16)s[i];
}

// ---------------------------------------------------------------------------
// Kernel 1: ball query + fused feature build  feat[c][p], c in {x,y,z,dx,dy,dz}
// ---------------------------------------------------------------------------
__global__ void k_fuse(const float* __restrict__ xyz, float* __restrict__ feat) {
  int t = blockIdx.x * blockDim.x + threadIdx.x;
  if (t >= BATCH * NPTS) return;
  int b = t / NPTS, n = t % NPTS;
  const float* base = xyz + (size_t)b * NPTS * 3;
  float cx = base[3*n], cy = base[3*n+1], cz = base[3*n+2];
  size_t p0 = (size_t)b * PPB_PRE + (size_t)n * NSAMP;
  int cnt = 0;
  float fx = 0.f, fy = 0.f, fz = 0.f;
  bool first = true;
  for (int j = 0; j < NPTS && cnt < NSAMP; ++j) {
    float dx = base[3*j]   - cx;
    float dy = base[3*j+1] - cy;
    float dz = base[3*j+2] - cz;
    float sq = dx*dx + dy*dy + dz*dz;
    if (sq <= RAD2) {                 // first NSAMP in-radius indices, in order
      if (first) { fx = dx; fy = dy; fz = dz; first = false; }
      size_t p = p0 + cnt;
      feat[0*PP+p] = cx; feat[1*PP+p] = cy; feat[2*PP+p] = cz;
      feat[3*PP+p] = dx; feat[4*PP+p] = dy; feat[5*PP+p] = dz;
      ++cnt;
    }
  }
  for (; cnt < NSAMP; ++cnt) {        // pad with first neighbor (always exists: j==n)
    size_t p = p0 + cnt;
    feat[0*PP+p] = cx; feat[1*PP+p] = cy; feat[2*PP+p] = cz;
    feat[3*PP+p] = fx; feat[4*PP+p] = fy; feat[5*PP+p] = fz;
  }
}

// ---------------------------------------------------------------------------
// Kernel 2: layer1 conv (K=6 too skinny for WMMA) -> raw h1 (f16), bias added
// ---------------------------------------------------------------------------
__global__ void k_conv1(const float* __restrict__ feat, const float* __restrict__ W,
                        const float* __restrict__ bias, _Float16* __restrict__ out) {
  int o = threadIdx.x;                       // 0..95
  size_t pbase = (size_t)blockIdx.x * 32;
  float w0=W[o*6+0], w1=W[o*6+1], w2=W[o*6+2], w3=W[o*6+3], w4=W[o*6+4], w5=W[o*6+5];
  float bi = bias[o];
  for (int i = 0; i < 32; ++i) {
    size_t p = pbase + i;
    float v = bi + w0*feat[0*PP+p] + w1*feat[1*PP+p] + w2*feat[2*PP+p]
                 + w3*feat[3*PP+p] + w4*feat[4*PP+p] + w5*feat[5*PP+p];
    out[(size_t)o*PP + p] = (_Float16)v;
  }
}

// ---------------------------------------------------------------------------
// GroupNorm statistics: deterministic two-stage reduction.
// Stage 1: grid (64 chunks, 16 (b,g) pairs); fixed-order LDS tree per block.
// ---------------------------------------------------------------------------
__global__ void k_stats(const _Float16* __restrict__ h, float* __restrict__ part,
                        int Cg, size_t PPB, size_t P) {
  int chunk = blockIdx.x;          // 0..63
  int bg    = blockIdx.y;          // 0..15
  int b = bg >> 3, g = bg & 7;
  size_t cnt = (size_t)Cg * PPB;
  size_t per = cnt / 64;           // always a multiple of 4
  size_t s0  = (size_t)chunk * per;
  float s1 = 0.f, s2 = 0.f;
  for (size_t i = s0 + (size_t)threadIdx.x * 4; i < s0 + per; i += 1024) {
    int    c = g * Cg + (int)(i / PPB);       // quad never crosses a row (PPB % 4 == 0)
    size_t p = (size_t)b * PPB + (i % PPB);
    v4h x = *(const v4h*)(h + (size_t)c * P + p);
#pragma unroll
    for (int q = 0; q < 4; ++q) { float v = (float)x[q]; s1 += v; s2 += v * v; }
  }
  __shared__ float r1[256], r2[256];
  r1[threadIdx.x] = s1; r2[threadIdx.x] = s2;
  __syncthreads();
  for (int o = 128; o > 0; o >>= 1) {
    if ((int)threadIdx.x < o) {
      r1[threadIdx.x] += r1[threadIdx.x + o];
      r2[threadIdx.x] += r2[threadIdx.x + o];
    }
    __syncthreads();
  }
  if (threadIdx.x == 0) {
    part[(bg * 64 + chunk) * 2 + 0] = r1[0];
    part[(bg * 64 + chunk) * 2 + 1] = r2[0];
  }
}

// Stage 2: sum 64 partials per (b,g) in fixed order -> mean, rstd
__global__ void k_stats_final(const float* __restrict__ part, float* __restrict__ mr,
                              float invCount) {
  int bg = threadIdx.x;
  if (bg >= 16) return;
  float s1 = 0.f, s2 = 0.f;
  for (int c = 0; c < 64; ++c) {
    s1 += part[(bg * 64 + c) * 2 + 0];
    s2 += part[(bg * 64 + c) * 2 + 1];
  }
  float m = s1 * invCount;
  float v = s2 * invCount - m * m;
  mr[bg * 2 + 0] = m;
  mr[bg * 2 + 1] = rsqrtf(v + 1e-5f);
}

// ---------------------------------------------------------------------------
// Generic WMMA conv1x1: Y(MxP) = Wh(MxK,f16) * GN_ReLU(X(KxP)) + bias
//   GN_IN : apply input-side GroupNorm+ReLU while staging tile into LDS
//   FINAL : last layer -> per-position L2 normalize and write f32 (B,N,96)
// LDS tile is stored POSITION-MAJOR (Xs[col][k], ld = K+8) so each lane's
// B fragment is 16 consecutive f16 -> two ds_load_b128. A fragment from
// row-major f16 weights is two runs of 8 consecutive f16 -> two b128 loads.
// ---------------------------------------------------------------------------
template<bool GN_IN, bool FINAL>
__global__ __launch_bounds__(256)
void k_wmma_conv(const _Float16* __restrict__ Xin, const _Float16* __restrict__ Wh,
                 const float* __restrict__ bias,
                 const float* __restrict__ gnMR, const float* __restrict__ gamma,
                 const float* __restrict__ beta,
                 _Float16* __restrict__ Yout, float* __restrict__ finalOut,
                 int M, int K, size_t P, size_t PPB) {
  __shared__ _Float16 Xs[64 * 200];        // position-major tile, ld = K+8 (pad)
  __shared__ float    Ybuf[96 * 64];       // FINAL only (eliminated otherwise)
  __shared__ float    nrm[64];             // FINAL only

  int tid = threadIdx.x;
  size_t pbase = (size_t)blockIdx.x * 64;
  int b   = (int)(pbase / PPB);
  int Cg  = K >> 3;                         // input channels per group
  int LDK = K + 8;

  // ---- stage input tile into LDS (transposed) with fused GroupNorm+ReLU --
  for (int idx = tid; idx < (K << 4); idx += 256) {   // K*16 quads of positions
    int c  = idx >> 4;
    int j4 = (idx & 15) << 2;
    const _Float16* src = Xin + (size_t)c * P + pbase + j4;
    if (j4 == 0) __builtin_prefetch(src + 64, 0, 0);  // next 64-position stripe
    v4h x = *(const v4h*)src;
    float sc = 1.f, sh = 0.f;
    if (GN_IN) {
      int g = c / Cg;
      float mean = gnMR[(b * 8 + g) * 2 + 0];
      float rstd = gnMR[(b * 8 + g) * 2 + 1];
      sc = rstd * gamma[c];
      sh = beta[c] - mean * sc;
    }
#pragma unroll
    for (int q = 0; q < 4; ++q) {
      float v = (float)x[q];
      if (GN_IN) v = fmaxf(0.f, v * sc + sh);
      Xs[(j4 + q) * LDK + c] = (_Float16)v;
    }
  }
  if (FINAL && tid < 64) nrm[tid] = 0.f;
  __syncthreads();

  // ---- WMMA tiles: (M/16) m-tiles x 4 n-tiles, round-robin over 8 waves --
  int wave = tid >> 5, lane = tid & 31;
  int lo16 = lane & 15;
  int h16  = (lane >> 4) << 3;              // 0 or 8
  int total = (M >> 4) << 2;
  for (int t = wave; t < total; t += 8) {
    int m0 = (t >> 2) << 4;
    int n0 = (t & 3) << 4;
    int mrow = m0 + lo16;
    int col  = n0 + lo16;
    const _Float16* wrow = Wh + (size_t)mrow * K + (h16 << 1);   // kg = 2*h16 (0 or 16), + k0
    const _Float16* xcol = Xs + col * LDK + (h16 << 1);   // kg = 2*h16 (0 or 16)
    v8f acc = {};
    for (int k0 = 0; k0 < K; k0 += 32) {
      v8h alo = *(const v8h*)(wrow + k0);
      v8h ahi = *(const v8h*)(wrow + k0 + 8);
      v16h a  = __builtin_shufflevector(alo, ahi, 0,1,2,3,4,5,6,7,8,9,10,11,12,13,14,15);
      v8h blo = *(const v8h*)(xcol + k0);
      v8h bhi = *(const v8h*)(xcol + k0 + 8);
      v16h bf = __builtin_shufflevector(blo, bhi, 0,1,2,3,4,5,6,7,8,9,10,11,12,13,14,15);
      acc = __builtin_amdgcn_wmma_f32_16x16x32_f16(
          false, a, false, bf, (short)0, acc, false, false);
    }
    // ---- epilogue ---------------------------------------------------------
#pragma unroll
    for (int r = 0; r < 8; ++r) {
      int row = m0 + r + h16;               // lanes 16-31 hold rows m0+8..m0+15
      float v = acc[r] + bias[row];
      if (FINAL) {
        atomicAdd(&nrm[col], v * v);
        Ybuf[row * 64 + col] = v;
      } else {
        Yout[(size_t)row * P + pbase + col] = (_Float16)v;
      }
    }
  }

  if (FINAL) {
    __syncthreads();
    for (int idx = tid; idx < M * 64; idx += 256) {
      int row = idx >> 6, col = idx & 63;
      size_t p = pbase + col;               // p = b*N + n  ->  out[(b,n,row)]
      finalOut[p * (size_t)M + row] = Ybuf[idx] * rsqrtf(nrm[col] + 1e-30f);
    }
  }
}

// ---------------------------------------------------------------------------
// Max-pool over samples with fused GN3 + ReLU:  pooled[c][b*N+n]
// ---------------------------------------------------------------------------
__global__ void k_pool(const _Float16* __restrict__ h3, const float* __restrict__ mr,
                       const float* __restrict__ gamma, const float* __restrict__ beta,
                       _Float16* __restrict__ pooled) {
  int t = blockIdx.x * blockDim.x + threadIdx.x;
  if (t >= 192 * (int)PP2) return;
  int c  = t / (int)PP2;
  int p2 = t % (int)PP2;
  int b  = p2 / NPTS;
  int g  = c / 24;
  float rstd = mr[(b * 8 + g) * 2 + 1];
  float sc = rstd * gamma[c];
  float sh = beta[c] - mr[(b * 8 + g) * 2 + 0] * sc;
  size_t base = (size_t)c * PP + (size_t)p2 * NSAMP;
  float m = 0.f;                            // post-ReLU values are >= 0
  for (int s = 0; s < NSAMP; s += 8) {
    v8h x = *(const v8h*)(h3 + base + s);
#pragma unroll
    for (int q = 0; q < 8; ++q)
      m = fmaxf(m, fmaxf(0.f, (float)x[q] * sc + sh));
  }
  pooled[(size_t)c * PP2 + p2] = (_Float16)m;
}

// ---------------------------------------------------------------------------
extern "C" void kernel_launch(void* const* d_in, const int* in_sizes, int n_in,
                              void* d_out, int out_size, void* d_ws, size_t ws_size,
                              hipStream_t stream) {
  const float* xyz     = (const float*)d_in[0];
  const float* pre_w1  = (const float*)d_in[1];
  const float* pre_b1  = (const float*)d_in[2];
  const float* pre_g1  = (const float*)d_in[3];
  const float* pre_be1 = (const float*)d_in[4];
  const float* pre_w2  = (const float*)d_in[5];
  const float* pre_b2  = (const float*)d_in[6];
  const float* pre_g2  = (const float*)d_in[7];
  const float* pre_be2 = (const float*)d_in[8];
  const float* pre_w3  = (const float*)d_in[9];
  const float* pre_b3  = (const float*)d_in[10];
  const float* pre_g3  = (const float*)d_in[11];
  const float* pre_be3 = (const float*)d_in[12];
  const float* post_w1 = (const float*)d_in[13];
  const float* post_b1 = (const float*)d_in[14];
  const float* post_g1 = (const float*)d_in[15];
  const float* post_be1= (const float*)d_in[16];
  const float* post_w2 = (const float*)d_in[17];
  const float* post_b2 = (const float*)d_in[18];
  const float* post_g2 = (const float*)d_in[19];
  const float* post_be2= (const float*)d_in[20];
  const float* post_w3 = (const float*)d_in[21];
  const float* post_b3 = (const float*)d_in[22];

  char* ws = (char*)d_ws;
  size_t off = 0;
  auto carve = [&](size_t bytes) {
    void* p = ws + off;
    off = (off + bytes + 255) & ~(size_t)255;
    return p;
  };
  float*    feat   = (float*)    carve(6   * PP  * 4);
  _Float16* h1     = (_Float16*) carve(96  * PP  * 2);
  _Float16* h2     = (_Float16*) carve(96  * PP  * 2);
  _Float16* h3     = (_Float16*) carve(192 * PP  * 2);
  _Float16* pooled = (_Float16*) carve(192 * PP2 * 2);
  _Float16* q1     = (_Float16*) carve(192 * PP2 * 2);
  _Float16* q2     = (_Float16*) carve(96  * PP2 * 2);
  float*    part   = (float*)    carve(16 * 64 * 2 * 4);
  float*    mr     = (float*)    carve(5 * 32 * 4);
  _Float16* wh2    = (_Float16*) carve(96  * 96  * 2);
  _Float16* wh3    = (_Float16*) carve(192 * 96  * 2);
  _Float16* wh4    = (_Float16*) carve(192 * 192 * 2);
  _Float16* wh5    = (_Float16*) carve(96  * 192 * 2);
  _Float16* wh6    = (_Float16*) carve(96  * 96  * 2);

  // one-time f16 weight conversion (tiny)
  k_cvt<<<(96*96  + 255) / 256, 256, 0, stream>>>(pre_w2,  wh2, 96*96);
  k_cvt<<<(192*96 + 255) / 256, 256, 0, stream>>>(pre_w3,  wh3, 192*96);
  k_cvt<<<(192*192+ 255) / 256, 256, 0, stream>>>(post_w1, wh4, 192*192);
  k_cvt<<<(96*192 + 255) / 256, 256, 0, stream>>>(post_w2, wh5, 96*192);
  k_cvt<<<(96*96  + 255) / 256, 256, 0, stream>>>(post_w3, wh6, 96*96);

  dim3 sgrid(64, 16);   // chunks x (b,g)

  k_fuse <<<(BATCH * NPTS + 255) / 256, 256, 0, stream>>>(xyz, feat);
  k_conv1<<<(int)(PP / 32), 96, 0, stream>>>(feat, pre_w1, pre_b1, h1);

  k_stats      <<<sgrid, 256, 0, stream>>>(h1, part, 12, PPB_PRE, PP);
  k_stats_final<<<1, 32, 0, stream>>>(part, mr + 0, 1.f / (12.f * PPB_PRE));
  k_wmma_conv<true,  false><<<(int)(PP / 64), 256, 0, stream>>>(
      h1, wh2, pre_b2, mr + 0, pre_g1, pre_be1, h2, nullptr, 96, 96, PP, PPB_PRE);

  k_stats      <<<sgrid, 256, 0, stream>>>(h2, part, 12, PPB_PRE, PP);
  k_stats_final<<<1, 32, 0, stream>>>(part, mr + 32, 1.f / (12.f * PPB_PRE));
  k_wmma_conv<true,  false><<<(int)(PP / 64), 256, 0, stream>>>(
      h2, wh3, pre_b3, mr + 32, pre_g2, pre_be2, h3, nullptr, 192, 96, PP, PPB_PRE);

  k_stats      <<<sgrid, 256, 0, stream>>>(h3, part, 24, PPB_PRE, PP);
  k_stats_final<<<1, 32, 0, stream>>>(part, mr + 64, 1.f / (24.f * PPB_PRE));
  k_pool<<<(int)((192 * PP2 + 255) / 256), 256, 0, stream>>>(h3, mr + 64, pre_g3, pre_be3, pooled);

  k_wmma_conv<false, false><<<(int)(PP2 / 64), 256, 0, stream>>>(
      pooled, wh4, post_b1, nullptr, nullptr, nullptr, q1, nullptr, 192, 192, PP2, (size_t)NPTS);

  k_stats      <<<sgrid, 256, 0, stream>>>(q1, part, 24, (size_t)NPTS, PP2);
  k_stats_final<<<1, 32, 0, stream>>>(part, mr + 96, 1.f / (24.f * NPTS));
  k_wmma_conv<true,  false><<<(int)(PP2 / 64), 256, 0, stream>>>(
      q1, wh5, post_b2, mr + 96, post_g1, post_be1, q2, nullptr, 96, 192, PP2, (size_t)NPTS);

  k_stats      <<<sgrid, 256, 0, stream>>>(q2, part, 12, (size_t)NPTS, PP2);
  k_stats_final<<<1, 32, 0, stream>>>(part, mr + 128, 1.f / (12.f * NPTS));
  k_wmma_conv<true,  true ><<<(int)(PP2 / 64), 256, 0, stream>>>(
      q2, wh6, post_b3, mr + 128, post_g2, post_be2, nullptr, (float*)d_out, 96, 96, PP2, (size_t)NPTS);
}
